// GNN_49967649521735
// MI455X (gfx1250) — matrix operand order repaired and checked by reading, not compiled, not run on hardware
//
#include <hip/hip_runtime.h>
#include <hip/hip_bf16.h>
#include <math.h>

// CDNA5 / gfx1250, wave32. bf16 WMMA GEMMs with fragment-order operands.

typedef __bf16 bf16_t;
typedef __attribute__((ext_vector_type(16))) __bf16 v16bf;
typedef __attribute__((ext_vector_type(8)))  __bf16 v8bf;
typedef __attribute__((ext_vector_type(4)))  __bf16 v4bf;
typedef __attribute__((ext_vector_type(8)))  float  v8f;

static __device__ __forceinline__ bf16_t f32_to_bf16_rne(float f) {
    union { float f; unsigned u; } x; x.f = f;
    unsigned r = x.u + 0x7FFFu + ((x.u >> 16) & 1u);   // round-to-nearest-even
    unsigned short h = (unsigned short)(r >> 16);
    bf16_t o; __builtin_memcpy(&o, &h, 2);
    return o;
}

// ---------------------------------------------------------------- utilities
__global__ void zero_f32(float* __restrict__ p, int n) {
    int i = blockIdx.x * blockDim.x + threadIdx.x;
    if (i < n) p[i] = 0.0f;
}

// ------------------------------------------------------------- degree/norms
__global__ void degree_kernel(const int* __restrict__ src, const int* __restrict__ dst,
                              float* __restrict__ outdeg, float* __restrict__ indeg, int E) {
    int e = blockIdx.x * blockDim.x + threadIdx.x;
    if (e < E) {
        atomicAdd(&outdeg[src[e]], 1.0f);
        atomicAdd(&indeg[dst[e]], 1.0f);
    }
}

__global__ void norm_finalize(float* __restrict__ deg, int n) {
    int i = blockIdx.x * blockDim.x + threadIdx.x;
    if (i < n) deg[i] = rsqrtf(fmaxf(deg[i], 1.0f));
}

// ------------------------------------------------- edge gather/scatter (SpMM)
__global__ void scatter_edges_f4(const float* __restrict__ h,
                                 const float* __restrict__ out_norm,
                                 const int* __restrict__ src,
                                 const int* __restrict__ dst,
                                 float* __restrict__ agg,
                                 int E, int F) {
    int idx = blockIdx.x * blockDim.x + threadIdx.x;
    int chunks = F >> 2;
    int e = idx / chunks;
    if (e >= E) return;
    int c = (idx - e * chunks) << 2;
    int s = src[e], d = dst[e];
    float ns = out_norm[s];
    float4 hv = *(const float4*)(h + (size_t)s * F + c);
    float* dp = agg + (size_t)d * F + c;
    atomicAdd(dp + 0, hv.x * ns);
    atomicAdd(dp + 1, hv.y * ns);
    atomicAdd(dp + 2, hv.z * ns);
    atomicAdd(dp + 3, hv.w * ns);
}

// ---------------------------- A prep: Abf[m,k] = bf16(agg[m,k] * in_norm[m])
__global__ void cvt_scale_rows_bf16(const float* __restrict__ agg,
                                    const float* __restrict__ rowscale,
                                    bf16_t* __restrict__ Abf,
                                    int M, int K) {
    int i4 = blockIdx.x * blockDim.x + threadIdx.x;
    long long base = (long long)i4 * 4;
    if (base >= (long long)M * K) return;
    int row = (int)(base / K);
    float rs = rowscale[row];
    float4 v = *(const float4*)(agg + base);
    v4bf o;
    o[0] = f32_to_bf16_rne(v.x * rs);
    o[1] = f32_to_bf16_rne(v.y * rs);
    o[2] = f32_to_bf16_rne(v.z * rs);
    o[3] = f32_to_bf16_rne(v.w * rs);
    *(v4bf*)(Abf + base) = o;
}

// -------- W prep: swizzle into WMMA B-fragment order [N/16][K/32][lane][16]
// lane's 16 elements = W[kb*32 + (lane>>4)*8 + (e&7) + (e>>3)*16, nt*16 + (lane&15)], e=0..15
__global__ void swizzle_W_bf16(const float* __restrict__ W,
                               bf16_t* __restrict__ Wsw,
                               int K, int N) {
    int t = blockIdx.x * blockDim.x + threadIdx.x;
    if (t >= K * N) return;
    int e    = t & 15;
    int l    = (t >> 4) & 31;
    int rest = t >> 9;
    int kblks = K >> 5;
    int kb = rest % kblks;
    int nt = rest / kblks;
    int k = (kb << 5) + ((l >> 4) << 3) + (e & 7) + ((e >> 3) << 4);
    int n = (nt << 4) + (l & 15);
    Wsw[t] = f32_to_bf16_rne(W[(size_t)k * N + n]);
}

// --------------------------------------------------------------- WMMA GEMM
// out[M,N] = act( Abf[M,K] @ W + bias ), W pre-swizzled to fragment order.
// One wave computes a 16x64 C strip (4 N-tiles) -> 4x A-fragment reuse.
__global__ __launch_bounds__(256) void gemm_bias_wmma(
    const bf16_t* __restrict__ Abf,
    const bf16_t* __restrict__ Wsw,
    const float* __restrict__ bias,
    float* __restrict__ out,
    int M, int K, int N, int relu)
{
    const int lane = threadIdx.x & 31;
    const int wave = threadIdx.x >> 5;
    const int NT4  = N >> 6;                 // strips of 4 N-tiles
    const int ntM  = (M + 15) >> 4;
    const int wq   = blockIdx.x * 8 + wave;
    if (wq >= ntM * NT4) return;             // wave-uniform exit

    const int mt = wq / NT4;
    const int ng = wq - mt * NT4;
    const int tM = mt << 4;
    const int nl   = lane & 15;
    const int half = lane >> 4;

    // accumulators: lane holds column nl of each 16x16 subtile
    v8f c0, c1, c2, c3;
    {
        const float* bb = bias + (ng << 6);
        float b0 = bb[nl], b1 = bb[16 + nl], b2 = bb[32 + nl], b3 = bb[48 + nl];
#pragma unroll
        for (int v = 0; v < 8; ++v) { c0[v] = b0; c1[v] = b1; c2[v] = b2; c3[v] = b3; }
    }

    // A: lane's row = tM + (lane&15), clamped branchlessly (M%16==0 normally)
    int rowA = tM + nl; if (rowA > M - 1) rowA = M - 1;
    const bf16_t* Arow = Abf + (size_t)rowA * K + (half << 3);

    const int kblks = K >> 5;
    const bf16_t* Wb = Wsw + (((size_t)(ng << 2) * kblks) << 9) + ((size_t)lane << 4);

    for (int kb = 0; kb < kblks; ++kb) {
        // A fragment: two contiguous 16B chunks (ascending K order)
        v8bf a0 = *(const v8bf*)(Arow + (kb << 5));
        v8bf a1 = *(const v8bf*)(Arow + (kb << 5) + 16);
        v16bf a = __builtin_shufflevector(a0, a1, 0,1,2,3,4,5,6,7,8,9,10,11,12,13,14,15);

        // B fragments: contiguous 32B per lane per subtile (pre-swizzled)
        const bf16_t* w0 = Wb + ((size_t)kb << 9);
        const size_t nstride = (size_t)kblks << 9;   // one N-tile of Wsw
#pragma unroll
        for (int j = 0; j < 4; ++j) {
            const bf16_t* wj = w0 + j * nstride;
            v8bf b0 = *(const v8bf*)(wj);
            v8bf b1 = *(const v8bf*)(wj + 8);
            v16bf b = __builtin_shufflevector(b0, b1, 0,1,2,3,4,5,6,7,8,9,10,11,12,13,14,15);
            v8f* cp = (j == 0) ? &c0 : (j == 1) ? &c1 : (j == 2) ? &c2 : &c3;
            *cp = __builtin_amdgcn_wmma_f32_16x16x32_bf16(
                      false, a, false, b, (short)0, *cp, false, false);
        }
    }

    // epilogue: wave-uniform full-tile fast path (M=50000 is 16-aligned)
    const int rbase = tM + (half << 3);
    float* op = out + (size_t)rbase * N + (ng << 6) + nl;
    if (tM + 16 <= M) {
#pragma unroll
        for (int v = 0; v < 8; ++v) {
            float* o = op + (size_t)v * N;
            float x0 = c0[v], x1 = c1[v], x2 = c2[v], x3 = c3[v];
            if (relu) {
                x0 = fmaxf(x0, 0.f); x1 = fmaxf(x1, 0.f);
                x2 = fmaxf(x2, 0.f); x3 = fmaxf(x3, 0.f);
            }
            o[0] = x0; o[16] = x1; o[32] = x2; o[48] = x3;
        }
    } else {
#pragma unroll
        for (int v = 0; v < 8; ++v) {
            if (rbase + v < M) {
                float* o = op + (size_t)v * N;
                float x0 = c0[v], x1 = c1[v], x2 = c2[v], x3 = c3[v];
                if (relu) {
                    x0 = fmaxf(x0, 0.f); x1 = fmaxf(x1, 0.f);
                    x2 = fmaxf(x2, 0.f); x3 = fmaxf(x3, 0.f);
                }
                o[0] = x0; o[16] = x1; o[32] = x2; o[48] = x3;
            }
        }
    }
}

// ----------------------------------------------- mean row-norm (wave reduce)
__global__ __launch_bounds__(256) void rownorm_kernel(const float* __restrict__ h,
                                                      float* __restrict__ sum,
                                                      int M, int N) {
    int gw   = (blockIdx.x * blockDim.x + threadIdx.x) >> 5;
    int lane = threadIdx.x & 31;
    if (gw >= M) return;
    const float* row = h + (size_t)gw * N;
    float s = 0.0f;
    for (int c = lane; c < N; c += 32) { float v = row[c]; s += v * v; }
#pragma unroll
    for (int off = 16; off > 0; off >>= 1) s += __shfl_xor(s, off, 32);
    if (lane == 0) atomicAdd(sum, sqrtf(s));
}

// ------------------------------------------------------ scaled sum pooling
__global__ void pool_kernel(const float* __restrict__ h,
                            const int* __restrict__ node2graph,
                            const float* __restrict__ sumnorm,
                            float* __restrict__ out, int M, int N) {
    int idx = blockIdx.x * blockDim.x + threadIdx.x;
    int chunks = N >> 2;
    int nIdx = idx / chunks;
    if (nIdx >= M) return;
    int c = (idx - nIdx * chunks) << 2;
    float factor = 16.0f * (float)M / sumnorm[0];   // sqrt(256)*M / sum(||h||)
    int g = node2graph[nIdx];
    float4 hv = *(const float4*)(h + (size_t)nIdx * N + c);
    float* op = out + (size_t)g * N + c;
    atomicAdd(op + 0, hv.x * factor);
    atomicAdd(op + 1, hv.y * factor);
    atomicAdd(op + 2, hv.z * factor);
    atomicAdd(op + 3, hv.w * factor);
}

// ---------------------------------------------------------------- launcher
extern "C" void kernel_launch(void* const* d_in, const int* in_sizes, int n_in,
                              void* d_out, int out_size, void* d_ws, size_t ws_size,
                              hipStream_t stream) {
    const float* x   = (const float*)d_in[0];
    const float* W1  = (const float*)d_in[1];
    const float* b1  = (const float*)d_in[2];
    const float* W2  = (const float*)d_in[3];
    const float* b2  = (const float*)d_in[4];
    const int*   src = (const int*)d_in[5];
    const int*   dst = (const int*)d_in[6];
    const int*   n2g = (const int*)d_in[7];
    float* out = (float*)d_out;

    const int E   = in_sizes[5];          // 800000
    const int N   = in_sizes[7];          // 50000 nodes
    const int FIN = in_sizes[0] / N;      // 128
    const int DIM = in_sizes[2];          // 256

    // workspace layout (floats)
    float* wsf     = (float*)d_ws;
    const int padN = (N + 63) & ~63;
    float* outnorm = wsf;                                   // [N]
    float* innorm  = wsf + padN;                            // [N]
    float* sumn    = wsf + 2 * (size_t)padN;                // [1]
    float* agg     = wsf + 2 * (size_t)padN + 64;           // [N x DIM]
    float* hbuf    = agg + (size_t)N * DIM;                 // [N x DIM]
    bf16_t* W1sw   = (bf16_t*)(hbuf + (size_t)N * DIM);     // [FIN x DIM] swizzled
    bf16_t* W2sw   = W1sw + (size_t)FIN * DIM;              // [DIM x DIM] swizzled
    bf16_t* Abf    = W2sw + (size_t)DIM * DIM;              // [N x DIM] bf16

    const int TB = 256;
    const int headFloats = 2 * padN + 64;

    zero_f32<<<(headFloats + TB - 1) / TB, TB, 0, stream>>>(wsf, headFloats);
    degree_kernel<<<(E + TB - 1) / TB, TB, 0, stream>>>(src, dst, outnorm, innorm, E);
    norm_finalize<<<(N + TB - 1) / TB, TB, 0, stream>>>(outnorm, N);
    norm_finalize<<<(N + TB - 1) / TB, TB, 0, stream>>>(innorm, N);
    swizzle_W_bf16<<<(FIN * DIM + TB - 1) / TB, TB, 0, stream>>>(W1, W1sw, FIN, DIM);
    swizzle_W_bf16<<<(DIM * DIM + TB - 1) / TB, TB, 0, stream>>>(W2, W2sw, DIM, DIM);

    // ------- layer 1: scatter 128-dim, bf16 convert+scale, WMMA GEMM + ReLU
    {
        long long nagg = (long long)N * FIN;
        zero_f32<<<(int)((nagg + TB - 1) / TB), TB, 0, stream>>>(agg, (int)nagg);
        long long thr = (long long)E * (FIN >> 2);
        scatter_edges_f4<<<(int)((thr + TB - 1) / TB), TB, 0, stream>>>(x, outnorm, src, dst, agg, E, FIN);
        long long q = nagg >> 2;
        cvt_scale_rows_bf16<<<(int)((q + TB - 1) / TB), TB, 0, stream>>>(agg, innorm, Abf, N, FIN);
        int strips = ((N + 15) >> 4) * (DIM >> 6);
        gemm_bias_wmma<<<(strips + 7) / 8, TB, 0, stream>>>(Abf, W1sw, b1, hbuf, N, FIN, DIM, 1);
    }
    // ------- layer 2: scatter 256-dim, convert+scale, WMMA GEMM (no act)
    {
        long long nagg = (long long)N * DIM;
        zero_f32<<<(int)((nagg + TB - 1) / TB), TB, 0, stream>>>(agg, (int)nagg);
        long long thr = (long long)E * (DIM >> 2);
        scatter_edges_f4<<<(int)((thr + TB - 1) / TB), TB, 0, stream>>>(hbuf, outnorm, src, dst, agg, E, DIM);
        long long q = nagg >> 2;
        cvt_scale_rows_bf16<<<(int)((q + TB - 1) / TB), TB, 0, stream>>>(agg, innorm, Abf, N, DIM);
        int strips = ((N + 15) >> 4) * (DIM >> 6);
        gemm_bias_wmma<<<(strips + 7) / 8, TB, 0, stream>>>(Abf, W2sw, b2, hbuf, N, DIM, DIM, 0);
    }

    // ------- mean L2 norm -> factor, per-graph sum pooling
    {
        rownorm_kernel<<<(N * 32 + TB - 1) / TB, TB, 0, stream>>>(hbuf, sumn, N, DIM);
        zero_f32<<<(out_size + TB - 1) / TB, TB, 0, stream>>>(out, out_size);
        long long thr = (long long)N * (DIM >> 2);
        pool_kernel<<<(int)((thr + TB - 1) / TB), TB, 0, stream>>>(hbuf, n2g, sumn, out, N, DIM);
    }
}
